// BiLSTMCRFModel_59846074302895
// MI455X (gfx1250) — matrix operand order repaired and checked by reading, not compiled, not run on hardware
//
#include <hip/hip_runtime.h>

// ---------------------------------------------------------------------------
// BiLSTM(2 layers, H=64, bidirectional) + FC(K=2) + CRF Viterbi decode
// Persistent-RNN style: one wave per 16-batch tile, WMMA bf16 for all GEMMs,
// TDM (tensor_load_to_lds) for weight staging, v_tanh_f32 activations.
// ---------------------------------------------------------------------------

typedef __bf16 bf16_t;
typedef __attribute__((ext_vector_type(16))) __bf16 v16bf;
typedef __attribute__((ext_vector_type(8)))  __bf16 v8bf;
typedef __attribute__((ext_vector_type(8)))  float  v8f;
typedef __attribute__((ext_vector_type(4)))  unsigned int u32x4;
typedef __attribute__((ext_vector_type(8)))  int    i32x8;
typedef __attribute__((ext_vector_type(4)))  int    i32x4;

static constexpr int T_LEN = 4096;
static constexpr int BATCH = 128;
static constexpr int HID   = 64;     // per-direction hidden
static constexpr int G4H   = 256;    // 4*HID gates

// workspace layout (bytes)
static constexpr size_t OFF_BHH0  = 0;                         // 2 dirs * 32KB
static constexpr size_t OFF_BHH1  = 65536;                     // 2 dirs * 32KB
static constexpr size_t OFF_BIH1  = 131072;                    // 2 dirs * 64KB
static constexpr size_t OFF_BSUM0 = 262144;                    // 2*256 f32
static constexpr size_t OFF_BSUM1 = 264192;                    // 2*256 f32
static constexpr size_t OFF_H0    = 266240;                    // A-frag layout, 128MB
static constexpr size_t SZ_H0     = (size_t)T_LEN*2*8*2*32*16*2;
static constexpr size_t OFF_H1    = OFF_H0 + SZ_H0;            // plain [t][b][128] bf16
static constexpr size_t SZ_H1     = (size_t)T_LEN*BATCH*128*2;
static constexpr size_t OFF_LOG   = OFF_H1 + SZ_H1;            // [t][b][2] f32
static constexpr size_t SZ_LOG    = (size_t)T_LEN*BATCH*2*4;
static constexpr size_t OFF_BP    = OFF_LOG + SZ_LOG;          // [t][b] u8

// ---------------- fast activations (hardware v_tanh_f32 on CDNA5) ----------
#if defined(__AMDGCN__) && __has_builtin(__builtin_amdgcn_tanhf)
__device__ __forceinline__ float fast_tanh(float x) { return __builtin_amdgcn_tanhf(x); }
#else
__device__ __forceinline__ float fast_tanh(float x) { return tanhf(x); }
#endif
__device__ __forceinline__ float fast_sigmoid(float x) {
    return 0.5f + 0.5f * fast_tanh(0.5f * x);
}

// ---------------- WMMA -----------------------------------------------------
__device__ __forceinline__ v8f wmma_bf16(v16bf a, v16bf b, v8f c) {
    // D = A(16x32) * B(32x16) + C, f32 accumulate
    return __builtin_amdgcn_wmma_f32_16x16x32_bf16(
        /*neg_a=*/false, a, /*neg_b=*/false, b,
        /*c_mod=*/(short)0, c, /*reuse_a=*/false, /*reuse_b=*/false);
}

union AFrag { v16bf v; v8bf h[2]; };

// ---------------- TDM: DMA a flat block (n8 x 8 bytes) global -> LDS -------
#if defined(__AMDGCN__) && __has_builtin(__builtin_amdgcn_tensor_load_to_lds) && \
    __has_builtin(__builtin_amdgcn_s_wait_tensorcnt)
#define HAVE_TDM 1
#endif

__device__ __forceinline__ void tdm_copy_to_lds(const void* gsrc, void* ldst,
                                                unsigned n8) {
#ifdef HAVE_TDM
    unsigned long long ga = (unsigned long long)gsrc;
    unsigned lds = (unsigned)(unsigned long long)ldst;   // low 32 bits = LDS offset
    // D# group 0: count=1 | lds_addr[63:32] | global_addr[120:64] | type=2
    u32x4 g0;
    g0.x = 1u;
    g0.y = lds;
    g0.z = (unsigned)ga;
    g0.w = ((unsigned)(ga >> 32) & 0x01FFFFFFu) | (2u << 30);
    // D# group 1: data_size=8B, tensor_dim0=n8, tensor_dim1=1, tile_dim0=n8,
    // tile_dim1=1, tensor_dim0_stride=n8
    i32x8 g1;
    g1[0] = (int)(3u << 16);                       // data_size = 3 (8 bytes)
    g1[1] = (int)((n8 & 0xFFFFu) << 16);           // tensor_dim0[15:0]
    g1[2] = (int)((n8 >> 16) | (1u << 16));        // tensor_dim0[31:16], tensor_dim1 = 1
    g1[3] = (int)((n8 & 0xFFFFu) << 16);           // tile_dim0 = n8
    g1[4] = 1;                                     // tile_dim1 = 1
    g1[5] = (int)n8;                               // tensor_dim0_stride[31:0]
    g1[6] = 0;
    g1[7] = 0;
    i32x4 z4 = {0, 0, 0, 0};
#if __clang_major__ >= 23
    i32x8 z8 = {0, 0, 0, 0, 0, 0, 0, 0};
    __builtin_amdgcn_tensor_load_to_lds(g0, g1, z4, z4, z8, 0);
#else
    __builtin_amdgcn_tensor_load_to_lds(g0, g1, z4, z4, 0);
#endif
#else
    (void)gsrc; (void)ldst; (void)n8;
#endif
}

__device__ __forceinline__ void tdm_wait() {
#ifdef HAVE_TDM
    __builtin_amdgcn_s_wait_tensorcnt(0);
#endif
}

// ---------------------------------------------------------------------------
// Pack a [G=256][Kdim] f32 weight matrix into WMMA B-fragment layout:
// out[kf][j][lane][16] bf16; tile (kf,j) covers K=[32kf,32kf+32), N=[16j,16j+16)
// ---------------------------------------------------------------------------
__global__ void pack_b(const float* __restrict__ W, bf16_t* __restrict__ out,
                       int Kdim, int nkf) {
    int id = blockIdx.x * blockDim.x + threadIdx.x;
    int total = nkf * 512;
    if (id >= total) return;
    int lane = id & 31;
    int j    = (id >> 5) & 15;
    int kf   = id >> 9;
    int n  = j * 16 + (lane & 15);
    int kb = kf * 32 + ((lane & 16) ? 16 : 0);
    #pragma unroll
    for (int e = 0; e < 16; ++e) {
        out[(size_t)id * 16 + e] = (bf16_t)W[(size_t)n * Kdim + (kb + e)];
    }
}

__global__ void bias_sum(const float* __restrict__ bi, const float* __restrict__ bh,
                         float* __restrict__ out, int n) {
    int id = blockIdx.x * blockDim.x + threadIdx.x;
    if (id < n) out[id] = bi[id] + bh[id];
}

// ---------------------------------------------------------------------------
// Layer 0 BiLSTM. grid.x = dir, 256 threads; wave w owns batches [16w,16w+16).
// Writes h0 in A-fragment layout: h0out[t][dir][m][kf][lane][16] bf16.
// ---------------------------------------------------------------------------
__global__ __launch_bounds__(256) void lstm_layer0(
    const float* __restrict__ x,        // [B][T][2]
    const float* __restrict__ w_ih0,    // [2][256][2]
    const bf16_t* __restrict__ bhh,     // packed [2][2][16][32][16]
    const float* __restrict__ bsum,     // [2][256]  (b_ih + b_hh)
    bf16_t* __restrict__ h0out)
{
    __shared__ bf16_t sB[2 * 16 * 32 * 16];  // 32KB: Whh^T B-frags for this dir
    __shared__ bf16_t stg[8 * 16 * HID];     // 16KB: per-wave h staging

    const int dir  = blockIdx.x;
    const int tid  = threadIdx.x;
    const int wave = tid >> 5;
    const int lane = tid & 31;
    const int col  = lane & 15;
    const int hi   = (lane >> 4) & 1;

#ifdef HAVE_TDM
    if (wave == 0) {
        tdm_copy_to_lds(bhh + (size_t)dir * 2 * 16 * 32 * 16, sB, 4096); // 32KB
        tdm_wait();
    }
#else
    {
        const u32x4* src = (const u32x4*)(bhh + (size_t)dir * 2 * 16 * 32 * 16);
        u32x4* dst = (u32x4*)sB;
        for (int i = tid; i < 2048; i += 256) dst[i] = src[i];
    }
#endif
    __syncthreads();

    // per-lane constants: input weights + bias for my 16 gate columns
    float wi0[16], wi1[16], bs[16];
    #pragma unroll
    for (int j = 0; j < 16; ++j) {
        int g = j * 16 + col;
        wi0[j] = w_ih0[((size_t)dir * G4H + g) * 2 + 0];
        wi1[j] = w_ih0[((size_t)dir * G4H + g) * 2 + 1];
        bs[j]  = bsum[dir * G4H + g];
    }

    float cst[4][8];
    #pragma unroll
    for (int jg = 0; jg < 4; ++jg)
        #pragma unroll
        for (int r = 0; r < 8; ++r) cst[jg][r] = 0.0f;

    AFrag a0, a1;
    #pragma unroll
    for (int e = 0; e < 16; ++e) { a0.v[e] = (bf16_t)0.0f; a1.v[e] = (bf16_t)0.0f; }

    const v16bf* Bf = (const v16bf*)sB;
    char* stgRow = (char*)&stg[(wave * 16 + col) * HID];   // A row = col
    const int o1 = hi ? 32 : 0;
    const int o2 = hi ? 48 : 16;

    for (int s = 0; s < T_LEN; ++s) {
        const int t = dir ? (T_LEN - 1 - s) : s;

        float x0v[8], x1v[8];
        #pragma unroll
        for (int r = 0; r < 8; ++r) {
            int b = wave * 16 + hi * 8 + r;
            const float* xp = x + ((size_t)b * T_LEN + t) * 2;
            x0v[r] = xp[0]; x1v[r] = xp[1];
        }

        #pragma unroll
        for (int jg = 0; jg < 4; ++jg) {
            v8f ai, af, ag, ao;
            #pragma unroll
            for (int r = 0; r < 8; ++r) {
                ai[r] = bs[jg]      + wi0[jg]      * x0v[r] + wi1[jg]      * x1v[r];
                af[r] = bs[jg + 4]  + wi0[jg + 4]  * x0v[r] + wi1[jg + 4]  * x1v[r];
                ag[r] = bs[jg + 8]  + wi0[jg + 8]  * x0v[r] + wi1[jg + 8]  * x1v[r];
                ao[r] = bs[jg + 12] + wi0[jg + 12] * x0v[r] + wi1[jg + 12] * x1v[r];
            }
            #pragma unroll
            for (int kf = 0; kf < 2; ++kf) {
                v16bf a = kf ? a1.v : a0.v;
                ai = wmma_bf16(a, Bf[(kf * 16 + jg     ) * 32 + lane], ai);
                af = wmma_bf16(a, Bf[(kf * 16 + jg + 4 ) * 32 + lane], af);
                ag = wmma_bf16(a, Bf[(kf * 16 + jg + 8 ) * 32 + lane], ag);
                ao = wmma_bf16(a, Bf[(kf * 16 + jg + 12) * 32 + lane], ao);
            }
            #pragma unroll
            for (int r = 0; r < 8; ++r) {
                float c = cst[jg][r];
                c = fast_sigmoid(af[r]) * c + fast_sigmoid(ai[r]) * fast_tanh(ag[r]);
                cst[jg][r] = c;
                float h = fast_sigmoid(ao[r]) * fast_tanh(c);
                int ml = hi * 8 + r;
                stg[(wave * 16 + ml) * HID + jg * 16 + col] = (bf16_t)h;
            }
        }

        // re-read own h as A fragments (wave-private LDS region)
        a0.h[0] = *(const v8bf*)(stgRow + o1);
        a0.h[1] = *(const v8bf*)(stgRow + o2);
        a1.h[0] = *(const v8bf*)(stgRow + 64 + o1);
        a1.h[1] = *(const v8bf*)(stgRow + 64 + o2);

        // store h(t) in A-frag layout for layer 1
        bf16_t* op = h0out + ((((size_t)t * 2 + dir) * 8 + wave) * 2) * (32 * 16);
        *(v16bf*)(op + (size_t)lane * 16)       = a0.v;
        *(v16bf*)(op + 512 + (size_t)lane * 16) = a1.v;
    }
}

// ---------------------------------------------------------------------------
// Layer 1 BiLSTM: input = concat(hf, hb) [128 feats]; K=128 input proj +
// K=64 recurrent proj, all WMMA. Dynamic LDS: 64K + 32K + 16K = 112KB.
// ---------------------------------------------------------------------------
__global__ __launch_bounds__(256) void lstm_layer1(
    const bf16_t* __restrict__ h0,      // [t][half][m][kf][lane][16]
    const bf16_t* __restrict__ bih,     // packed [2][4][16][32][16]
    const bf16_t* __restrict__ bhh,     // packed [2][2][16][32][16]
    const float* __restrict__ bsum,     // [2][256]
    bf16_t* __restrict__ h1)            // plain [t][b][128]
{
    extern __shared__ char smem[];
    bf16_t* sBI = (bf16_t*)smem;                 // 4*16*32*16  (64KB)
    bf16_t* sBH = sBI + 4 * 16 * 32 * 16;        // 2*16*32*16  (32KB)
    bf16_t* stg = sBH + 2 * 16 * 32 * 16;        // 8*16*64     (16KB)

    const int dir  = blockIdx.x;
    const int tid  = threadIdx.x;
    const int wave = tid >> 5;
    const int lane = tid & 31;
    const int col  = lane & 15;
    const int hi   = (lane >> 4) & 1;

#ifdef HAVE_TDM
    if (wave == 0) {
        tdm_copy_to_lds(bih + (size_t)dir * 4 * 16 * 32 * 16, sBI, 8192); // 64KB
        tdm_copy_to_lds(bhh + (size_t)dir * 2 * 16 * 32 * 16, sBH, 4096); // 32KB
        tdm_wait();
    }
#else
    {
        const u32x4* s1 = (const u32x4*)(bih + (size_t)dir * 4 * 16 * 32 * 16);
        u32x4* d1 = (u32x4*)sBI;
        for (int i = tid; i < 4096; i += 256) d1[i] = s1[i];
        const u32x4* s2 = (const u32x4*)(bhh + (size_t)dir * 2 * 16 * 32 * 16);
        u32x4* d2 = (u32x4*)sBH;
        for (int i = tid; i < 2048; i += 256) d2[i] = s2[i];
    }
#endif
    __syncthreads();

    float bs[16];
    #pragma unroll
    for (int j = 0; j < 16; ++j) bs[j] = bsum[dir * G4H + j * 16 + col];

    float cst[4][8];
    #pragma unroll
    for (int jg = 0; jg < 4; ++jg)
        #pragma unroll
        for (int r = 0; r < 8; ++r) cst[jg][r] = 0.0f;

    AFrag ah0, ah1;
    #pragma unroll
    for (int e = 0; e < 16; ++e) { ah0.v[e] = (bf16_t)0.0f; ah1.v[e] = (bf16_t)0.0f; }

    const v16bf* BI = (const v16bf*)sBI;
    const v16bf* BH = (const v16bf*)sBH;
    char* stgRow = (char*)&stg[(wave * 16 + col) * HID];
    const int o1 = hi ? 32 : 0;
    const int o2 = hi ? 48 : 16;

    for (int s = 0; s < T_LEN; ++s) {
        const int t = dir ? (T_LEN - 1 - s) : s;

        // input A fragments: p covers feature K=[32p,32p+32); half=p>>1, kf=p&1
        AFrag ax[4];
        #pragma unroll
        for (int p = 0; p < 4; ++p) {
            size_t idx = (((((size_t)t * 2 + (p >> 1)) * 8 + wave) * 2 + (p & 1)) * 32 + lane) * 16;
            ax[p].v = *(const v16bf*)(h0 + idx);
        }
        if (s + 1 < T_LEN) {
            int tn = dir ? (T_LEN - 2 - s) : (s + 1);
            __builtin_prefetch(h0 + ((((size_t)tn * 2) * 8 + wave) * 2 * 32) * 16, 0, 0);
        }

        #pragma unroll
        for (int jg = 0; jg < 4; ++jg) {
            v8f ai, af, ag, ao;
            #pragma unroll
            for (int r = 0; r < 8; ++r) {
                ai[r] = bs[jg]; af[r] = bs[jg + 4]; ag[r] = bs[jg + 8]; ao[r] = bs[jg + 12];
            }
            #pragma unroll
            for (int p = 0; p < 4; ++p) {
                ai = wmma_bf16(ax[p].v, BI[(p * 16 + jg     ) * 32 + lane], ai);
                af = wmma_bf16(ax[p].v, BI[(p * 16 + jg + 4 ) * 32 + lane], af);
                ag = wmma_bf16(ax[p].v, BI[(p * 16 + jg + 8 ) * 32 + lane], ag);
                ao = wmma_bf16(ax[p].v, BI[(p * 16 + jg + 12) * 32 + lane], ao);
            }
            #pragma unroll
            for (int kf = 0; kf < 2; ++kf) {
                v16bf a = kf ? ah1.v : ah0.v;
                ai = wmma_bf16(a, BH[(kf * 16 + jg     ) * 32 + lane], ai);
                af = wmma_bf16(a, BH[(kf * 16 + jg + 4 ) * 32 + lane], af);
                ag = wmma_bf16(a, BH[(kf * 16 + jg + 8 ) * 32 + lane], ag);
                ao = wmma_bf16(a, BH[(kf * 16 + jg + 12) * 32 + lane], ao);
            }
            #pragma unroll
            for (int r = 0; r < 8; ++r) {
                float c = cst[jg][r];
                c = fast_sigmoid(af[r]) * c + fast_sigmoid(ai[r]) * fast_tanh(ag[r]);
                cst[jg][r] = c;
                float h = fast_sigmoid(ao[r]) * fast_tanh(c);
                int ml = hi * 8 + r;
                int b = wave * 16 + ml;
                int hh = jg * 16 + col;
                stg[(wave * 16 + ml) * HID + hh] = (bf16_t)h;
                h1[((size_t)t * BATCH + b) * 128 + dir * HID + hh] = (bf16_t)h;
            }
        }

        ah0.h[0] = *(const v8bf*)(stgRow + o1);
        ah0.h[1] = *(const v8bf*)(stgRow + o2);
        ah1.h[0] = *(const v8bf*)(stgRow + 64 + o1);
        ah1.h[1] = *(const v8bf*)(stgRow + 64 + o2);
    }
}

// ---------------------------------------------------------------------------
// FC: logits[t][b][k] = h1[t][b][:] . fc_w[k][:] + fc_b[k]   (K=2)
// ---------------------------------------------------------------------------
__global__ void fc_kernel(const bf16_t* __restrict__ h1,
                          const float* __restrict__ fcw, const float* __restrict__ fcb,
                          float* __restrict__ logits) {
    int idx = blockIdx.x * blockDim.x + threadIdx.x;   // t*B + b
    if (idx >= T_LEN * BATCH) return;
    const bf16_t* hp = h1 + (size_t)idx * 128;
    float s0 = fcb[0], s1 = fcb[1];
    #pragma unroll 8
    for (int h = 0; h < 128; ++h) {
        float v = (float)hp[h];
        s0 += v * fcw[h];
        s1 += v * fcw[128 + h];
    }
    logits[(size_t)idx * 2 + 0] = s0;
    logits[(size_t)idx * 2 + 1] = s1;
}

// ---------------------------------------------------------------------------
// Viterbi decode, K=2. One lane per batch element.
// ---------------------------------------------------------------------------
__global__ void viterbi_kernel(const float* __restrict__ logits,
                               const float* __restrict__ trans,
                               const float* __restrict__ startp,
                               const float* __restrict__ endp,
                               unsigned char* __restrict__ bp,
                               int* __restrict__ out) {
    int b = threadIdx.x;
    if (b >= BATCH) return;
    const float t00 = trans[0], t01 = trans[1], t10 = trans[2], t11 = trans[3];
    float s0 = startp[0] + logits[(size_t)b * 2 + 0];
    float s1 = startp[1] + logits[(size_t)b * 2 + 1];
    for (int t = 1; t < T_LEN; ++t) {
        float c00 = s0 + t00, c10 = s1 + t10;   // -> tag 0
        float c01 = s0 + t01, c11 = s1 + t11;   // -> tag 1
        int b0 = c10 > c00; float m0 = b0 ? c10 : c00;
        int b1 = c11 > c01; float m1 = b1 ? c11 : c01;
        bp[(size_t)(t - 1) * BATCH + b] = (unsigned char)(b0 | (b1 << 1));
        const float* lp = logits + ((size_t)t * BATCH + b) * 2;
        s0 = m0 + lp[0];
        s1 = m1 + lp[1];
    }
    int tag = (s1 + endp[1] > s0 + endp[0]) ? 1 : 0;
    out[(size_t)b * T_LEN + (T_LEN - 1)] = tag;
    for (int t = T_LEN - 2; t >= 0; --t) {
        tag = (bp[(size_t)t * BATCH + b] >> tag) & 1;
        out[(size_t)b * T_LEN + t] = tag;
    }
}

// ---------------------------------------------------------------------------
extern "C" void kernel_launch(void* const* d_in, const int* in_sizes, int n_in,
                              void* d_out, int out_size, void* d_ws, size_t ws_size,
                              hipStream_t stream) {
    (void)in_sizes; (void)n_in; (void)out_size; (void)ws_size;

    const float* x     = (const float*)d_in[0];
    const float* w_ih0 = (const float*)d_in[1];
    const float* w_hh0 = (const float*)d_in[2];
    const float* b_ih0 = (const float*)d_in[3];
    const float* b_hh0 = (const float*)d_in[4];
    const float* w_ih1 = (const float*)d_in[5];
    const float* w_hh1 = (const float*)d_in[6];
    const float* b_ih1 = (const float*)d_in[7];
    const float* b_hh1 = (const float*)d_in[8];
    const float* fc_w  = (const float*)d_in[9];
    const float* fc_b  = (const float*)d_in[10];
    const float* trans = (const float*)d_in[11];
    const float* start = (const float*)d_in[12];
    const float* endp  = (const float*)d_in[13];

    char* ws = (char*)d_ws;
    bf16_t* bhh0  = (bf16_t*)(ws + OFF_BHH0);
    bf16_t* bhh1  = (bf16_t*)(ws + OFF_BHH1);
    bf16_t* bih1  = (bf16_t*)(ws + OFF_BIH1);
    float*  bsum0 = (float*)(ws + OFF_BSUM0);
    float*  bsum1 = (float*)(ws + OFF_BSUM1);
    bf16_t* h0    = (bf16_t*)(ws + OFF_H0);
    bf16_t* h1    = (bf16_t*)(ws + OFF_H1);
    float*  logi  = (float*)(ws + OFF_LOG);
    unsigned char* bp = (unsigned char*)(ws + OFF_BP);

    for (int dir = 0; dir < 2; ++dir) {
        pack_b<<<4, 256, 0, stream>>>(w_hh0 + (size_t)dir * G4H * HID,
                                      bhh0 + (size_t)dir * 16384, HID, 2);
        pack_b<<<4, 256, 0, stream>>>(w_hh1 + (size_t)dir * G4H * HID,
                                      bhh1 + (size_t)dir * 16384, HID, 2);
        pack_b<<<8, 256, 0, stream>>>(w_ih1 + (size_t)dir * G4H * 128,
                                      bih1 + (size_t)dir * 32768, 128, 4);
    }
    bias_sum<<<2, 256, 0, stream>>>(b_ih0, b_hh0, bsum0, 512);
    bias_sum<<<2, 256, 0, stream>>>(b_ih1, b_hh1, bsum1, 512);

    lstm_layer0<<<2, 256, 0, stream>>>(x, w_ih0, bhh0, bsum0, h0);
    lstm_layer1<<<2, 256, 114688, stream>>>(h0, bih1, bhh1, bsum1, h1);

    fc_kernel<<<(T_LEN * BATCH) / 256, 256, 0, stream>>>(h1, fc_w, fc_b, logi);
    viterbi_kernel<<<1, 128, 0, stream>>>(logi, trans, start, endp, bp, (int*)d_out);
}
